// SelfAttentionHead_28484223107149
// MI455X (gfx1250) — compile-verified, matrix operand order checked by reading, not run on hardware
//
#include <hip/hip_runtime.h>

// ---------------------------------------------------------------------------
// MI455X (gfx1250) fused single-head attention, bf16 WMMA + flash softmax.
//   B=4, S=4096, D=1024, Hd=128, fp32 in/out.
// ---------------------------------------------------------------------------

typedef __attribute__((ext_vector_type(16))) __bf16 v16bf;
typedef __attribute__((ext_vector_type(2)))  __bf16 v2bf;
typedef __attribute__((ext_vector_type(2)))  float  v2f;
typedef __attribute__((ext_vector_type(8)))  float  v8f;

union FragB { v16bf v; uint4 q[2]; unsigned u[8]; };   // 16x32 A or 32x16 B bf16 fragment
union AccF  { v8f   v; float f[8]; };                  // 16x16 f32 C/D fragment

// f32x2 -> packed bf16x2 via IR fptrunc: backend picks native cvt if available.
__device__ __forceinline__ unsigned pack2bf(float a, float b) {
    v2f  f = {a, b};
    v2bf p = __builtin_convertvector(f, v2bf);
    unsigned r;
    __builtin_memcpy(&r, &p, 4);
    return r;
}
__device__ __forceinline__ unsigned short f2bf(float f) {
    __bf16 h = (__bf16)f;
    unsigned short r;
    __builtin_memcpy(&r, &h, 2);
    return r;
}

#define WMMA_BF16(A, B, C) \
    __builtin_amdgcn_wmma_f32_16x16x32_bf16(false, (A), false, (B), (short)0, (C), false, false)

// ---------------------------------------------------------------------------
// Kernel 0a: x [16384,1024] fp32 -> xb bf16 (vectorized, memory-bound).
// 8 elements per thread: 2x b128 load, 1x b128 store.
// ---------------------------------------------------------------------------
__global__ __launch_bounds__(256) void k_cvt(const float* __restrict__ x,
                                             unsigned short* __restrict__ xb) {
    size_t i = ((size_t)blockIdx.x * 256 + threadIdx.x) * 8;
    float4 f0 = *(const float4*)(const void*)(x + i);
    float4 f1 = *(const float4*)(const void*)(x + i + 4);
    uint4 o;
    o.x = pack2bf(f0.x, f0.y);
    o.y = pack2bf(f0.z, f0.w);
    o.z = pack2bf(f1.x, f1.y);
    o.w = pack2bf(f1.z, f1.w);
    *(uint4*)(void*)(xb + i) = o;
}

// ---------------------------------------------------------------------------
// Kernel 0b: W [D=1024, Hd=128] fp32 -> Wt [Hd=128, D=1024] bf16 (x3 matrices)
// ---------------------------------------------------------------------------
__global__ __launch_bounds__(256) void k_wt(const float* __restrict__ Wq,
                                            const float* __restrict__ Wk,
                                            const float* __restrict__ Wv,
                                            unsigned short* __restrict__ Wt) {
    int tid = blockIdx.x * 256 + threadIdx.x;      // 0 .. 3*131072-1
    int mat = tid >> 17;
    int rem = tid & 131071;
    int h   = rem >> 10;
    int d   = rem & 1023;
    const float* W = (mat == 0) ? Wq : ((mat == 1) ? Wk : Wv);
    Wt[tid] = f2bf(W[d * 128 + h]);
}

// ---------------------------------------------------------------------------
// Kernel 1: QKV projection.  [16384,1024] x [1024,128] per matrix, all bf16 in.
// One wave = one 16-row tile x all 8 col tiles; inner loop is pure
// global_load_b128 + v_wmma (A conversion was hoisted into k_cvt).
// Q,K stored row-major bf16; V stored transposed [B,128,4096] bf16.
// ---------------------------------------------------------------------------
__global__ __launch_bounds__(256) void k_proj(const unsigned short* __restrict__ xb,
                                              const unsigned short* __restrict__ Wt,
                                              unsigned short* __restrict__ Qb,
                                              unsigned short* __restrict__ Kb,
                                              unsigned short* __restrict__ Vt) {
    const int lane = threadIdx.x & 31;
    const int w    = threadIdx.x >> 5;
    const int half = lane >> 4;
    const int ln   = lane & 15;
    const int mat  = blockIdx.y;
    const int grow0 = (blockIdx.x * 8 + w) * 16;

    const unsigned short* Wm = Wt + (size_t)mat * (128 * 1024);

    AccF c[8];
    #pragma unroll
    for (int t = 0; t < 8; ++t)
        #pragma unroll
        for (int i = 0; i < 8; ++i) c[t].f[i] = 0.f;

    const unsigned short* xrow = xb + (size_t)(grow0 + ln) * 1024;
    for (int kk = 0; kk < 1024; kk += 32) {
        FragB a;   // A: x row (grow0+ln), bf16 chunks at +8*half and +16+8*half
        a.q[0] = *(const uint4*)(const void*)(xrow + kk + 8 * half);
        a.q[1] = *(const uint4*)(const void*)(xrow + kk + 16 + 8 * half);
        #pragma unroll
        for (int t = 0; t < 8; ++t) {
            FragB bfr;   // B col n = Wt row (t*16+ln): contiguous bf16
            const unsigned short* wr = Wm + (size_t)(t * 16 + ln) * 1024 + kk;
            bfr.q[0] = *(const uint4*)(const void*)(wr + 8 * half);
            bfr.q[1] = *(const uint4*)(const void*)(wr + 16 + 8 * half);
            c[t].v = WMMA_BF16(a.v, bfr.v, c[t].v);
        }
    }

    if (mat < 2) {
        unsigned short* O = (mat == 0) ? Qb : Kb;
        #pragma unroll
        for (int t = 0; t < 8; ++t)
            #pragma unroll
            for (int r = 0; r < 8; ++r) {
                int row = grow0 + r + 8 * half;
                O[(size_t)row * 128 + t * 16 + ln] = f2bf(c[t].f[r]);
            }
    } else {  // V transposed: Vt[b][h][s]
        #pragma unroll
        for (int t = 0; t < 8; ++t)
            #pragma unroll
            for (int r = 0; r < 8; ++r) {
                int srow = grow0 + r + 8 * half;
                int b = srow >> 12, s = srow & 4095;
                Vt[((size_t)(b * 128 + t * 16 + ln)) * 4096 + s] = f2bf(c[t].f[r]);
            }
    }
}

// ---------------------------------------------------------------------------
// Kernel 2: flash attention. One wave = one 16-query tile; key tiles of 32.
// P (16x32 f32 C-layout) is staged through a private 1KB LDS tile to become
// a bf16 A-layout fragment for the PV matmul; s_wait_dscnt gives cross-lane
// visibility (DS ops are in-order per wave, so no workgroup barrier needed).
// ---------------------------------------------------------------------------
__global__ __launch_bounds__(256) void k_attn(const unsigned short* __restrict__ Qb,
                                              const unsigned short* __restrict__ Kb,
                                              const unsigned short* __restrict__ Vt,
                                              float* __restrict__ out) {
    __shared__ __align__(16) unsigned short Pld[8 * 16 * 32];  // 8 KB, 1 KB per wave
    const int lane = threadIdx.x & 31;
    const int w    = threadIdx.x >> 5;
    const int half = lane >> 4;
    const int ln   = lane & 15;
    const int q0   = (blockIdx.x * 8 + w) * 16;
    const int b    = q0 >> 12;
    const int kb   = b << 12;
    const float scale = 0.08838834764831845f;  // 1/sqrt(128)

    // Q resident in registers: 4 A-fragments covering Hd=128
    FragB qf[4];
    {
        const unsigned short* qr = Qb + (size_t)(q0 + ln) * 128;
        #pragma unroll
        for (int ff = 0; ff < 4; ++ff) {
            qf[ff].q[0] = *(const uint4*)(const void*)(qr + 32 * ff + 8 * half);
            qf[ff].q[1] = *(const uint4*)(const void*)(qr + 32 * ff + 16 + 8 * half);
        }
    }

    AccF o[8];
    #pragma unroll
    for (int t = 0; t < 8; ++t)
        #pragma unroll
        for (int i = 0; i < 8; ++i) o[t].f[i] = 0.f;
    float mrow[8], lrow[8];
    #pragma unroll
    for (int r = 0; r < 8; ++r) { mrow[r] = -1e30f; lrow[r] = 0.f; }

    unsigned short* myP = &Pld[w * 512];

    for (int kt = 0; kt < 4096; kt += 32) {
        AccF s0, s1;
        #pragma unroll
        for (int i = 0; i < 8; ++i) { s0.f[i] = 0.f; s1.f[i] = 0.f; }

        // S = Q * K^T : B col n = K row (kt+n) -> contiguous bf16 loads
        const unsigned short* kr0 = Kb + (size_t)(kb + kt + ln) * 128;
        const unsigned short* kr1 = kr0 + 16 * 128;
        #pragma unroll
        for (int ff = 0; ff < 4; ++ff) {
            FragB kf, kg;
            kf.q[0] = *(const uint4*)(const void*)(kr0 + 32 * ff + 8 * half);
            kf.q[1] = *(const uint4*)(const void*)(kr0 + 32 * ff + 16 + 8 * half);
            s0.v = WMMA_BF16(qf[ff].v, kf.v, s0.v);
            kg.q[0] = *(const uint4*)(const void*)(kr1 + 32 * ff + 8 * half);
            kg.q[1] = *(const uint4*)(const void*)(kr1 + 32 * ff + 16 + 8 * half);
            s1.v = WMMA_BF16(qf[ff].v, kg.v, s1.v);
        }

        // Online softmax. C layout: row = r + 8*half, col = ln. Row reductions
        // with xor masks 8..1 stay within each 16-lane half -> exactly one row.
        #pragma unroll
        for (int r = 0; r < 8; ++r) {
            float a  = s0.f[r] * scale;
            float bb = s1.f[r] * scale;
            float mt = fmaxf(a, bb);
            #pragma unroll
            for (int off = 8; off >= 1; off >>= 1)
                mt = fmaxf(mt, __shfl_xor(mt, off, 32));
            float mnew  = fmaxf(mrow[r], mt);
            float alpha = __expf(mrow[r] - mnew);
            float p0 = __expf(a - mnew);
            float p1 = __expf(bb - mnew);
            float ps = p0 + p1;
            #pragma unroll
            for (int off = 8; off >= 1; off >>= 1)
                ps += __shfl_xor(ps, off, 32);
            lrow[r] = lrow[r] * alpha + ps;
            mrow[r] = mnew;
            #pragma unroll
            for (int t = 0; t < 8; ++t) o[t].f[r] *= alpha;   // rescale O row r
            int row = r + 8 * half;
            myP[row * 32 + ln]      = f2bf(p0);               // stage P row-major 16x32
            myP[row * 32 + 16 + ln] = f2bf(p1);
        }

        asm volatile("s_wait_dscnt 0" ::: "memory");          // cross-lane LDS visibility
        FragB pf;                                             // P as 16x32 A fragment
        pf.q[0] = *(const uint4*)(const void*)(myP + ln * 32 + 8 * half);
        pf.q[1] = *(const uint4*)(const void*)(myP + ln * 32 + 16 + 8 * half);
        asm volatile("" ::: "memory");                        // keep next writes after reads

        // O += P * V : B col n = Vt row (b*128 + t*16 + n) -> contiguous bf16
        #pragma unroll
        for (int t = 0; t < 8; ++t) {
            FragB vf;
            const unsigned short* vr = Vt + (size_t)(b * 128 + t * 16 + ln) * 4096 + kt;
            vf.q[0] = *(const uint4*)(const void*)(vr + 8 * half);
            vf.q[1] = *(const uint4*)(const void*)(vr + 16 + 8 * half);
            o[t].v = WMMA_BF16(pf.v, vf.v, o[t].v);
        }
    }

    // Normalize by row sums and store fp32
    #pragma unroll
    for (int r = 0; r < 8; ++r) {
        float inv = __builtin_amdgcn_rcpf(lrow[r]);
        int row = q0 + r + 8 * half;
        #pragma unroll
        for (int t = 0; t < 8; ++t)
            out[(size_t)row * 128 + t * 16 + ln] = o[t].f[r] * inv;
    }
}

// ---------------------------------------------------------------------------
extern "C" void kernel_launch(void* const* d_in, const int* in_sizes, int n_in,
                              void* d_out, int out_size, void* d_ws, size_t ws_size,
                              hipStream_t stream) {
    const float* x  = (const float*)d_in[0];
    const float* Wq = (const float*)d_in[1];
    const float* Wk = (const float*)d_in[2];
    const float* Wv = (const float*)d_in[3];
    float* out = (float*)d_out;

    // workspace partition (bf16 halves), all 16B-aligned:
    unsigned short* ws = (unsigned short*)d_ws;
    unsigned short* Wt = ws;                          //  3 * 128*1024   (768 KB)
    unsigned short* Qb = Wt + 3 * 131072;             // 16384*128       (4 MB)
    unsigned short* Kb = Qb + 16384 * 128;            // 16384*128       (4 MB)
    unsigned short* Vt = Kb + 16384 * 128;            // [B][128][4096]  (4 MB)
    unsigned short* xb = Vt + 16384 * 128;            // 16384*1024      (33.5 MB)

    k_cvt <<<dim3(8192),   dim3(256), 0, stream>>>(x, xb);         // 16M elems / 8 per thread
    k_wt  <<<dim3(1536),   dim3(256), 0, stream>>>(Wq, Wk, Wv, Wt);
    k_proj<<<dim3(128, 3), dim3(256), 0, stream>>>(xb, Wt, Qb, Kb, Vt);
    k_attn<<<dim3(128),    dim3(256), 0, stream>>>(Qb, Kb, Vt, out);
}
